// SetAbstraction_34050500722878
// MI455X (gfx1250) — matrix operand order repaired and checked by reading, not compiled, not run on hardware
//
#include <hip/hip_runtime.h>
#include <hip/hip_bf16.h>
#include <cstdint>
#include <cstddef>

typedef __attribute__((ext_vector_type(16))) _Float16 v16h;
typedef __attribute__((ext_vector_type(8)))  _Float16 h8;
typedef __attribute__((ext_vector_type(8)))  float    v8f;

#define B_    4
#define N_    8192
#define S_    2048
#define CIN   64
#define CMID  64
#define COUT  128
#define KPAD1 96          // 67 padded to 96 (3 x K32 WMMA steps)
#define KPAD2 64
#define NSAMP 32
#define R2_   0.04f       // RADIUS^2
#define EPS_  1e-5f
#define NEWP_ELEMS (B_*S_*3)

// ---- workspace layout (bytes) ----
#define WS_IDX   0                          // int  [B*S*32]        = 1048576 B
#define WS_W1T   (WS_IDX + B_*S_*NSAMP*4)   // f16  [64*96]         = 12288 B
#define WS_B1    (WS_W1T + CMID*KPAD1*2)    // f32  [64]
#define WS_W2T   (WS_B1  + CMID*4)          // f16  [128*64]        = 16384 B
#define WS_B2    (WS_W2T + COUT*KPAD2*2)    // f32  [128]

union V16u { v16h v; h8 h[2]; };

// A-operand (16x32 f16) per ISA 7.12.2: lane g=lane>>4 holds row (lane&15),
// K in two runs of 8: [k0+8g .. +7] and [k0+16+8g .. +7]
__device__ __forceinline__ v16h lds_load_A(const _Float16* base, int kpad,
                                           int row0, int k0, int lane) {
  const int g = lane >> 4, r = lane & 15;
  const _Float16* p = base + (row0 + r) * kpad + k0 + 8 * g;
  V16u u;
  u.h[0] = *(const h8*)(p);
  u.h[1] = *(const h8*)(p + 16);
  return u.v;
}

// B-operand (32x16 f16): lane holds column (lane&15),
// K in two runs of 8: [k0+8g .. +7] and [k0+16+8g .. +7] (weights staged [n][k]).
__device__ __forceinline__ v16h lds_load_B(const _Float16* base, int kpad,
                                           int n0, int k0, int lane) {
  const int g = lane >> 4, nl = lane & 15;
  const _Float16* p = base + (n0 + nl) * kpad + k0 + 8 * g;
  V16u u;
  u.h[0] = *(const h8*)(p);
  u.h[1] = *(const h8*)(p + 16);
  return u.v;
}

// ---------------- kernel 1: gather new_p ----------------
__global__ void gather_newp(const float* __restrict__ p,
                            const int* __restrict__ sample_idx,
                            float* __restrict__ out_np) {
  int q = blockIdx.x * blockDim.x + threadIdx.x;
  if (q >= B_ * S_) return;
  int b = q >> 11;                 // / S_
  int s = q & (S_ - 1);
  int i = sample_idx[b * S_ + s];
  const float* src = p + ((size_t)b * N_ + i) * 3;
  float* dst = out_np + (size_t)q * 3;
  dst[0] = src[0]; dst[1] = src[1]; dst[2] = src[2];
}

// ---------------- kernel 2: ball query (wave32 per query) ----------------
// "first 32 in-radius indices, ascending, pad with first" == reference top_k trick.
__global__ void ball_query(const float* __restrict__ p,
                           const float* __restrict__ newp,   // d_out part 1
                           int* __restrict__ idxbuf) {
  const int lane = threadIdx.x & 31;
  const int wv   = threadIdx.x >> 5;
  const int qg   = blockIdx.x * 8 + wv;          // 8 waves per block
  if (qg >= B_ * S_) return;
  const int b = qg >> 11;
  const float qx = newp[qg * 3 + 0];
  const float qy = newp[qg * 3 + 1];
  const float qz = newp[qg * 3 + 2];
  int* oq = idxbuf + (size_t)qg * NSAMP;
  const float* pb = p + (size_t)b * N_ * 3;

  int cnt = 0;
  for (int base = 0; base < N_ && cnt < NSAMP; base += 32) {
    const int i = base + lane;
    const float dx = pb[i * 3 + 0] - qx;
    const float dy = pb[i * 3 + 1] - qy;
    const float dz = pb[i * 3 + 2] - qz;
    const bool hit = (dx * dx + dy * dy + dz * dz) < R2_;
    const unsigned mask = (unsigned)__ballot(hit);
    const int rank = __popc(mask & ((1u << lane) - 1u));
    if (hit && (cnt + rank) < NSAMP) oq[cnt + rank] = i;
    cnt += __popc(mask);
  }
  // the query point itself is always in radius -> cnt >= 1; pad tail with first
  if (lane >= cnt && lane < NSAMP) {
    int first = oq[0];
    oq[lane] = first;
  }
}

// ---------------- kernel 3: fold BN into f16 weights ----------------
__global__ void prep_weights(const float* __restrict__ w1,   // (64,67)
                             const float* __restrict__ g1, const float* __restrict__ b1,
                             const float* __restrict__ m1, const float* __restrict__ v1,
                             const float* __restrict__ w2,   // (128,64)
                             const float* __restrict__ g2, const float* __restrict__ b2,
                             const float* __restrict__ m2, const float* __restrict__ v2,
                             _Float16* __restrict__ w1t, float* __restrict__ bias1,
                             _Float16* __restrict__ w2t, float* __restrict__ bias2) {
  const int tid = threadIdx.x;
  for (int e = tid; e < CMID * KPAD1; e += blockDim.x) {
    int n = e / KPAD1, k = e - n * KPAD1;
    float s = g1[n] * rsqrtf(v1[n] + EPS_);
    w1t[e] = (k < 67) ? (_Float16)(w1[n * 67 + k] * s) : (_Float16)0.0f;
  }
  if (tid < CMID) {
    float s = g1[tid] * rsqrtf(v1[tid] + EPS_);
    bias1[tid] = b1[tid] - m1[tid] * s;
  }
  for (int e = tid; e < COUT * KPAD2; e += blockDim.x) {
    int n = e >> 6;
    float s = g2[n] * rsqrtf(v2[n] + EPS_);
    w2t[e] = (_Float16)(w2[e] * s);                 // (128,64) already [n][k]
  }
  if (tid < COUT) {
    float s = g2[tid] * rsqrtf(v2[tid] + EPS_);
    bias2[tid] = b2[tid] - m2[tid] * s;
  }
}

// ---------------- kernel 4: fused grouping + 2x WMMA MLP + top3 pool ----------
// 128 threads = 4 waves; block handles 2 queries = 64 rows (4 M-tiles of 16).
__global__ void __launch_bounds__(128)
fused_mlp_pool(const float* __restrict__ p,
               const float* __restrict__ f,          // (B,64,N)
               const int* __restrict__ idxbuf,
               const float* __restrict__ newp,       // d_out part 1
               const _Float16* __restrict__ w1t, const float* __restrict__ bias1,
               const _Float16* __restrict__ w2t, const float* __restrict__ bias2,
               float* __restrict__ out_pooled) {     // d_out + NEWP_ELEMS
  __shared__ _Float16 sW1t[CMID * KPAD1];   // 12 KB  [n][k], BN-folded
  __shared__ float    sB1[CMID];
  __shared__ _Float16 sW2t[COUT * KPAD2];   // 16 KB
  __shared__ float    sB2[COUT];
  __shared__ _Float16 sFeat[64 * KPAD1];    // 12 KB  [row][k], zero-padded K
  __shared__ _Float16 sH1[64 * KPAD2];      //  8 KB  [row][k]
  __shared__ float    sH2[64 * 16];         //  4 KB  per-N-tile pooling staging
  __shared__ int      sIdx[64];
  __shared__ float    sNp[6];

  const int tid  = threadIdx.x;
  const int wv   = tid >> 5;                // M-tile owned by this wave
  const int lane = tid & 31;
  const int g    = lane >> 4;
  const int nl   = lane & 15;
  const int qg0  = blockIdx.x * 2;          // first global query of this block
  const int b    = qg0 >> 11;

  // ---- stage weights / indices / query points ----
  for (int e = tid; e < CMID * KPAD1; e += 128) sW1t[e] = w1t[e];
  for (int e = tid; e < COUT * KPAD2; e += 128) sW2t[e] = w2t[e];
  if (tid < CMID) sB1[tid] = bias1[tid];
  if (tid < COUT) sB2[tid] = bias2[tid];
  if (tid < 64)   sIdx[tid] = idxbuf[(size_t)qg0 * NSAMP + tid];
  if (tid < 6)    sNp[tid] = newp[qg0 * 3 + tid];
  __syncthreads();

  // ---- stage feat: rows = (local query q, neighbor j), cols = [dp(3) | f(64) | 0-pad] ----
  for (int e = tid; e < KPAD1 * 64; e += 128) {
    const int k = e >> 6;                   // channel-major: neighbor gathers coalesce
    const int r = e & 63;
    const int i = sIdx[r];
    float val;
    if (k < 3)       val = p[((size_t)b * N_ + i) * 3 + k] - sNp[(r >> 5) * 3 + k];
    else if (k < 67) val = f[((size_t)b * CIN + (k - 3)) * N_ + i];
    else             val = 0.0f;
    sFeat[r * KPAD1 + k] = (_Float16)val;
  }
  __syncthreads();

  // ---- layer 1: (64 x 96) @ (96 x 64), bias in C accumulator, relu -> sH1 ----
  for (int n = 0; n < 4; ++n) {
    const float bz = sB1[16 * n + nl];
    v8f c;
#pragma unroll
    for (int v = 0; v < 8; ++v) c[v] = bz;
#pragma unroll
    for (int kk = 0; kk < 3; ++kk) {
      v16h a  = lds_load_A(sFeat, KPAD1, 16 * wv, kk * 32, lane);
      v16h bm = lds_load_B(sW1t,  KPAD1, 16 * n,  kk * 32, lane);
      c = __builtin_amdgcn_wmma_f32_16x16x32_f16(false, a, false, bm,
                                                 (short)0, c, false, false);
    }
#pragma unroll
    for (int v = 0; v < 8; ++v) {
      float x = fmaxf(c[v], 0.0f);
      sH1[(16 * wv + v + 8 * g) * KPAD2 + 16 * n + nl] = (_Float16)x;
    }
  }
  __syncthreads();

  // ---- layer 2: (64 x 64) @ (64 x 128), relu, then top-3 pool per (query, channel) ----
  for (int n = 0; n < 8; ++n) {
    const float bz = sB2[16 * n + nl];
    v8f c;
#pragma unroll
    for (int v = 0; v < 8; ++v) c[v] = bz;
#pragma unroll
    for (int kk = 0; kk < 2; ++kk) {
      v16h a  = lds_load_A(sH1,  KPAD2, 16 * wv, kk * 32, lane);
      v16h bm = lds_load_B(sW2t, KPAD2, 16 * n,  kk * 32, lane);
      c = __builtin_amdgcn_wmma_f32_16x16x32_f16(false, a, false, bm,
                                                 (short)0, c, false, false);
    }
#pragma unroll
    for (int v = 0; v < 8; ++v)
      sH2[(16 * wv + v + 8 * g) * 16 + nl] = fmaxf(c[v], 0.0f);
    __syncthreads();

    if (tid < 32) {                          // 2 queries x 16 channels of this N-tile
      const int q = tid >> 4, cc = tid & 15;
      float t0 = -1.0f, t1 = -1.0f, t2 = -1.0f;   // values are >= 0 after relu
#pragma unroll
      for (int j = 0; j < 32; ++j) {
        float x = sH2[(q * 32 + j) * 16 + cc];
        if (x > t0)      { t2 = t1; t1 = t0; t0 = x; }
        else if (x > t1) { t2 = t1; t1 = x; }
        else if (x > t2) { t2 = x; }
      }
      const int qg = qg0 + q;
      const int bb = qg >> 11, s = qg & (S_ - 1);
      const int cglob = 16 * n + cc;
      out_pooled[((size_t)bb * COUT + cglob) * S_ + s] = t0 + 0.25f * (t1 + t2);
    }
    __syncthreads();
  }
}

extern "C" void kernel_launch(void* const* d_in, const int* in_sizes, int n_in,
                              void* d_out, int out_size, void* d_ws, size_t ws_size,
                              hipStream_t stream) {
  const float* p        = (const float*)d_in[0];
  const float* f        = (const float*)d_in[1];
  const float* conv1_w  = (const float*)d_in[2];
  const float* bn1_g    = (const float*)d_in[3];
  const float* bn1_b    = (const float*)d_in[4];
  const float* bn1_m    = (const float*)d_in[5];
  const float* bn1_v    = (const float*)d_in[6];
  const float* conv2_w  = (const float*)d_in[7];
  const float* bn2_g    = (const float*)d_in[8];
  const float* bn2_b    = (const float*)d_in[9];
  const float* bn2_m    = (const float*)d_in[10];
  const float* bn2_v    = (const float*)d_in[11];
  const int*   sample_i = (const int*)d_in[12];

  float* out_np     = (float*)d_out;
  float* out_pooled = out_np + NEWP_ELEMS;

  char* ws = (char*)d_ws;
  int*      idxbuf = (int*)(ws + WS_IDX);
  _Float16* w1t    = (_Float16*)(ws + WS_W1T);
  float*    bias1  = (float*)(ws + WS_B1);
  _Float16* w2t    = (_Float16*)(ws + WS_W2T);
  float*    bias2  = (float*)(ws + WS_B2);

  gather_newp<<<(B_ * S_ + 255) / 256, 256, 0, stream>>>(p, sample_i, out_np);
  ball_query<<<(B_ * S_) / 8, 256, 0, stream>>>(p, out_np, idxbuf);
  prep_weights<<<1, 256, 0, stream>>>(conv1_w, bn1_g, bn1_b, bn1_m, bn1_v,
                                      conv2_w, bn2_g, bn2_b, bn2_m, bn2_v,
                                      w1t, bias1, w2t, bias2);
  fused_mlp_pool<<<(B_ * S_) / 2, 128, 0, stream>>>(p, f, idxbuf, out_np,
                                                    w1t, bias1, w2t, bias2,
                                                    out_pooled);
}
